// GATConv_10737418240426
// MI455X (gfx1250) — matrix operand order repaired and checked by reading, not compiled, not run on hardware
//
#include <hip/hip_runtime.h>
#include <hip/hip_bf16.h>

typedef __attribute__((ext_vector_type(16))) __bf16 v16bf;
typedef __attribute__((ext_vector_type(8)))  float  v8f;

#define FDIM 128
#define NEG_SLOPE 0.2f

// ---------- helpers ----------
__device__ __forceinline__ unsigned f2ord(float f) {
  unsigned u = __float_as_uint(f);
  return (u & 0x80000000u) ? ~u : (u | 0x80000000u);
}
__device__ __forceinline__ float ord2f(unsigned u) {
  return (u & 0x80000000u) ? __uint_as_float(u & 0x7FFFFFFFu)
                           : __uint_as_float(~u);
}
#define ORD_NEG_INF 0x007FFFFFu   // f2ord(-inf)

// ---------- K0: init output accumulator + segment max/sum ----------
__global__ __launch_bounds__(256) void gat_init(float* __restrict__ out,
                                                unsigned* __restrict__ mmax,
                                                float* __restrict__ ssum,
                                                int n_nodes, int total) {
  int t = blockIdx.x * blockDim.x + threadIdx.x;
  if (t < total) out[t] = 0.0f;
  if (t < n_nodes) { mmax[t] = ORD_NEG_INF; ssum[t] = 0.0f; }
}

// ---------- K1: Wh = h @ W via bf16 WMMA ----------
// One block (8 wave32) per 16-row tile; wave w computes 16x16 tile at col 16*w.
__global__ __launch_bounds__(256) void gat_wh_gemm(const float* __restrict__ h,
                                                   const float* __restrict__ W,
                                                   float* __restrict__ Wh,
                                                   int n_nodes) {
  __shared__ __bf16 As[16][FDIM];   // 4 KB staged A tile (bf16)

  const int r0 = blockIdx.x * 16;
  const int t  = threadIdx.x;

  // Cooperative stage of the 16x128 A tile (each thread: 8 contiguous floats)
  {
    int idx = t * 8;               // 0..2040
    int r   = idx >> 7;
    int c   = idx & (FDIM - 1);
    int row = r0 + r;
    if (row >= n_nodes) row = n_nodes - 1;   // clamp (value unused on store)
    const float4* src = (const float4*)(h + (size_t)row * FDIM + c);
    float4 v0 = src[0];
    float4 v1 = src[1];
    __bf16* dst = &As[r][c];
    dst[0] = (__bf16)v0.x; dst[1] = (__bf16)v0.y;
    dst[2] = (__bf16)v0.z; dst[3] = (__bf16)v0.w;
    dst[4] = (__bf16)v1.x; dst[5] = (__bf16)v1.y;
    dst[6] = (__bf16)v1.z; dst[7] = (__bf16)v1.w;
  }
  __syncthreads();

  const int wave = t >> 5;
  const int lane = t & 31;
  const int n0   = wave * 16;      // output column tile
  const int nn   = lane & 15;
  const int half = lane >> 4;      // 0 or 1
  const int m    = lane & 15;      // A-matrix row for this lane

  v8f acc = {};
#pragma unroll
  for (int k0 = 0; k0 < FDIM; k0 += 32) {
    // A fragment (16-bit A 16x32): K contiguous per half-wave, k = half*16 + {0..15}
    v16bf afrag, bfrag;
    const int ka = k0 + half * 16;
#pragma unroll
    for (int e = 0; e < 8; ++e) {
      afrag[e]     = As[m][ka + e];
      afrag[e + 8] = As[m][ka + 8 + e];
    }
    // B fragment (dense 32x16 bf16): N = lane&15, K contiguous per half-wave
    const int kb = k0 + half * 16;
#pragma unroll
    for (int e = 0; e < 16; ++e) {
      bfrag[e] = (__bf16)W[(size_t)(kb + e) * FDIM + (n0 + nn)];
    }
    acc = __builtin_amdgcn_wmma_f32_16x16x32_bf16(
        /*neg_a=*/false, afrag, /*neg_b=*/false, bfrag,
        /*c_mod=*/(short)0, acc, /*reuse_a=*/false, /*reuse_b=*/false);
  }

  // C/D layout: VGPR r -> M = r + half*8 ; N = lane&15
#pragma unroll
  for (int r = 0; r < 8; ++r) {
    int row = r0 + half * 8 + r;
    if (row < n_nodes) Wh[(size_t)row * FDIM + n0 + nn] = acc[r];
  }
}

// ---------- K2: alpha_src / alpha_dst (one wave per row) ----------
__global__ __launch_bounds__(256) void gat_alpha(const float* __restrict__ Wh,
                                                 const float* __restrict__ a,
                                                 float* __restrict__ asrc,
                                                 float* __restrict__ adst,
                                                 int n_nodes) {
  const int wave = threadIdx.x >> 5;
  const int lane = threadIdx.x & 31;
  const int row  = blockIdx.x * 8 + wave;
  if (row >= n_nodes) return;
  const float4 wv = *(const float4*)(Wh + (size_t)row * FDIM + lane * 4);
  const float4 a1 = *(const float4*)(a + lane * 4);
  const float4 a2 = *(const float4*)(a + FDIM + lane * 4);
  float s1 = wv.x * a1.x + wv.y * a1.y + wv.z * a1.z + wv.w * a1.w;
  float s2 = wv.x * a2.x + wv.y * a2.y + wv.z * a2.z + wv.w * a2.w;
#pragma unroll
  for (int off = 16; off; off >>= 1) {
    s1 += __shfl_xor(s1, off, 32);
    s2 += __shfl_xor(s2, off, 32);
  }
  if (lane == 0) { asrc[row] = s1; adst[row] = s2; }
}

// ---------- K3: edge logits + segment max ----------
__global__ __launch_bounds__(256) void gat_edge_logits(const int* __restrict__ rowi,
                                                       const int* __restrict__ coli,
                                                       const float* __restrict__ asrc,
                                                       const float* __restrict__ adst,
                                                       float* __restrict__ ee,
                                                       unsigned* __restrict__ mmax,
                                                       int n_edges) {
  int e = blockIdx.x * blockDim.x + threadIdx.x;
  if (e >= n_edges) return;
  int r = rowi[e], c = coli[e];
  float v = asrc[r] + adst[c];
  v = (v > 0.0f) ? v : NEG_SLOPE * v;       // leaky_relu
  ee[e] = v;
  atomicMax(&mmax[c], f2ord(v));
}

// ---------- K4: exp(e - max) + segment sum ----------
__global__ __launch_bounds__(256) void gat_edge_exp(const int* __restrict__ coli,
                                                    float* __restrict__ ee,
                                                    const unsigned* __restrict__ mmax,
                                                    float* __restrict__ ssum,
                                                    int n_edges) {
  int e = blockIdx.x * blockDim.x + threadIdx.x;
  if (e >= n_edges) return;
  int c = coli[e];
  float m  = ord2f(mmax[c]);
  float ex = __expf(ee[e] - m);
  ee[e] = ex;
  atomicAdd(&ssum[c], ex);
}

// ---------- K5: attended scatter-add (one wave per edge) ----------
__global__ __launch_bounds__(256) void gat_aggregate(const int* __restrict__ rowi,
                                                     const int* __restrict__ coli,
                                                     const float* __restrict__ ee,
                                                     const float* __restrict__ ssum,
                                                     const float* __restrict__ Wh,
                                                     float* __restrict__ out,
                                                     int n_edges) {
  const int wave = threadIdx.x >> 5;
  const int lane = threadIdx.x & 31;
  const int e    = blockIdx.x * 8 + wave;
  if (e >= n_edges) return;
  int r = rowi[e], c = coli[e];
  float attn = ee[e] / (ssum[c] + 1e-16f);
  const float4 wv = *(const float4*)(Wh + (size_t)r * FDIM + lane * 4);
  float* dst = out + (size_t)c * FDIM + lane * 4;
  atomicAdd(dst + 0, wv.x * attn);
  atomicAdd(dst + 1, wv.y * attn);
  atomicAdd(dst + 2, wv.z * attn);
  atomicAdd(dst + 3, wv.w * attn);
}

// ---------- K6: ELU in place ----------
__global__ __launch_bounds__(256) void gat_elu(float* __restrict__ out, int total) {
  int t = blockIdx.x * blockDim.x + threadIdx.x;
  if (t >= total) return;
  float x = out[t];
  out[t] = (x > 0.0f) ? x : (__expf(x) - 1.0f);
}

extern "C" void kernel_launch(void* const* d_in, const int* in_sizes, int n_in,
                              void* d_out, int out_size, void* d_ws, size_t ws_size,
                              hipStream_t stream) {
  const float* h  = (const float*)d_in[0];
  const int*   ei = (const int*)d_in[1];     // [2, E] (row; col)
  const float* W  = (const float*)d_in[2];   // [128,128] row-major (K x N)
  const float* a  = (const float*)d_in[3];   // [256]

  const int n_nodes = in_sizes[0] / FDIM;
  const int n_edges = in_sizes[1] / 2;
  const int total   = n_nodes * FDIM;

  const int* rowi = ei;
  const int* coli = ei + n_edges;

  // Workspace layout
  float*    Wh   = (float*)d_ws;                       // n_nodes*128
  float*    asrc = Wh + (size_t)n_nodes * FDIM;        // n_nodes
  float*    adst = asrc + n_nodes;                     // n_nodes
  float*    ssum = adst + n_nodes;                     // n_nodes
  unsigned* mmax = (unsigned*)(ssum + n_nodes);        // n_nodes
  float*    ee   = (float*)(mmax + n_nodes);           // n_edges

  float* out = (float*)d_out;

  gat_init<<<(total + 255) / 256, 256, 0, stream>>>(out, mmax, ssum, n_nodes, total);

  gat_wh_gemm<<<(n_nodes + 15) / 16, 256, 0, stream>>>(h, W, Wh, n_nodes);

  gat_alpha<<<(n_nodes + 7) / 8, 256, 0, stream>>>(Wh, a, asrc, adst, n_nodes);

  gat_edge_logits<<<(n_edges + 255) / 256, 256, 0, stream>>>(rowi, coli, asrc, adst,
                                                             ee, mmax, n_edges);

  gat_edge_exp<<<(n_edges + 255) / 256, 256, 0, stream>>>(coli, ee, mmax, ssum, n_edges);

  gat_aggregate<<<(n_edges + 7) / 8, 256, 0, stream>>>(rowi, coli, ee, ssum, Wh,
                                                       out, n_edges);

  gat_elu<<<(total + 255) / 256, 256, 0, stream>>>(out, total);
}
